// MultiQueryAttention_81870666596846
// MI455X (gfx1250) — matrix operand order repaired and checked by reading, not compiled, not run on hardware
//
#include <hip/hip_runtime.h>
#include <hip/hip_bf16.h>

// ---------------------------------------------------------------------------
// MultiQueryAttention for MI455X (gfx1250), bf16 WMMA pipeline with
// TDM (tensor_load_to_lds) double-buffered GEMM.
//   E=1024, H=16, D=64, Q=4, B=4, S=2048, M=B*S=8192
// ---------------------------------------------------------------------------

typedef __attribute__((ext_vector_type(16))) __bf16       v16bf;
typedef __attribute__((ext_vector_type(8)))  float        v8f;
typedef __attribute__((ext_vector_type(4)))  int          v4i;
typedef __attribute__((ext_vector_type(4)))  unsigned int u32x4;
typedef __attribute__((ext_vector_type(8)))  int          i32x8;
typedef __attribute__((ext_vector_type(4)))  int          i32x4;

union Frag16 {            // 32 bytes: one WMMA 16-bit A/B operand per lane
  v16bf  v;
  v4i    q[2];
  __bf16 e[16];
};

#define WMMA_BF16(a, b, c) \
  __builtin_amdgcn_wmma_f32_16x16x32_bf16(false, (a), false, (b), (short)0, (c), false, false)

static constexpr int kB = 4, kS = 2048, kE = 1024, kH = 16, kD = 64, kQ = 4;
static constexpr int kM = kB * kS;  // 8192 rows (tokens)

// ---------------------------------------------------------------------------
// TDM: issue one 2D tile load (tile_rows x tile_cols bf16, row-major, row
// stride = row_stride elements) from global into LDS at byte offset ldsOff.
// D# per CDNA5 ISA ch.8: group0 = {count/lds_addr/global_addr/type},
// group1 = {data_size, tensor dims, tile dims, dim0 stride}, groups 2/3 = 0
// (2D tile: tile_dim2/3/4 = 0).  6-arg builtin form (clang-23 toolchain).
// ---------------------------------------------------------------------------
__device__ __forceinline__ void tdm_load_2d(unsigned ldsOff, const void* gptr,
                                            int tile_cols, int tile_rows,
                                            int row_stride, int td0, int td1) {
  unsigned long long ga = (unsigned long long)(uintptr_t)gptr;
  u32x4 g0;
  g0[0] = 1u;                                         // count=1, user descriptor
  g0[1] = ldsOff;                                     // lds_addr (bytes)
  g0[2] = (unsigned)(ga & 0xFFFFFFFFull);             // global_addr[31:0]
  g0[3] = (unsigned)((ga >> 32) & 0x1FFFFFFull)       // global_addr[56:32]
          | (2u << 30);                               // type = 2 ("image")
  i32x8 g1;
  g1[0] = (1 << 16);                                  // data_size=1 (2B), mask=0
  g1[1] = (int)((unsigned)(td0 & 0xFFFF) << 16);      // tensor_dim0[15:0]
  g1[2] = (int)(((unsigned)(td0 >> 16) & 0xFFFF)      // tensor_dim0[31:16]
          | ((unsigned)(td1 & 0xFFFF) << 16));        // tensor_dim1[15:0]
  g1[3] = (int)(((unsigned)(td1 >> 16) & 0xFFFF)      // tensor_dim1[31:16]
          | ((unsigned)tile_cols << 16));             // tile_dim0
  g1[4] = tile_rows;                                  // tile_dim1 (tile_dim2=0)
  g1[5] = row_stride;                                 // tensor_dim0_stride[31:0]
  g1[6] = 0;                                          // stride0 hi / stride1 lo
  g1[7] = 0;
  i32x4 z4 = {0, 0, 0, 0};
  i32x8 z8 = {0, 0, 0, 0, 0, 0, 0, 0};
  __builtin_amdgcn_tensor_load_to_lds(g0, g1, z4, z4, z8, 0);
}

// ---------------------------------------------------------------------------
// fp32 -> bf16 conversion (4 elements / thread)
// ---------------------------------------------------------------------------
__global__ __launch_bounds__(256) void cvt_f32_bf16(const float* __restrict__ in,
                                                    __bf16* __restrict__ out, int n) {
  int i = (blockIdx.x * 256 + threadIdx.x) * 4;
  if (i + 3 < n) {
    out[i + 0] = (__bf16)in[i + 0];
    out[i + 1] = (__bf16)in[i + 1];
    out[i + 2] = (__bf16)in[i + 2];
    out[i + 3] = (__bf16)in[i + 3];
  }
}

// ---------------------------------------------------------------------------
// Tiled WMMA GEMM:  out(m,n) = sum_k A[m,k] * W[n,k] + bias[n]
//   A: M x K bf16 row-major, W: N x K bf16 row-major.
//   BM=128, BN=128, BK=32 tiles, double-buffered in LDS and fed by the TDM
//   (wave 0 issues tensor_load_to_lds, waits TENSORcnt, barrier releases the
//   7 other waves; wave 0 computes too).  8 waves, each 64x32 (4x2 WMMA 16x16 tiles).
//   The single static __shared__ array sits at group-segment offset 0, so the
//   TDM D# lds_addr uses literal byte offsets 0/8192/16384/24576.
//   Output mapping (stacked-Wq support):
//     off = (n / n_per_group)*group_stride + m*n_per_group + (n % n_per_group)
// ---------------------------------------------------------------------------
template <bool OUT_F32>
__global__ __launch_bounds__(256) void gemm_bf16_tn(
    const __bf16* __restrict__ A, const __bf16* __restrict__ W,
    const float* __restrict__ bias, void* __restrict__ outv,
    int M, int N, int K, int n_per_group, long long group_stride) {
  // [A0 | A1 | B0 | B1], each 128x32 bf16 = 8KB  => 32KB total, offset 0.
  __shared__ __align__(16) __bf16 lds[4 * 128 * 32];

  const int t    = threadIdx.x;
  const int wave = t >> 5;
  const int lane = t & 31;
  const int lh   = lane >> 4;   // K-half selector (A/B operand layout)
  const int ll   = lane & 15;   // row (A) / column (B) within tile
  const int wm   = wave >> 2;   // 0..1
  const int wn   = wave & 3;    // 0..3
  const int tileM = blockIdx.y * 128;
  const int tileN = blockIdx.x * 128;

  const __bf16* Abase = A + (size_t)tileM * K;
  const __bf16* Wbase = W + (size_t)tileN * K;
  const int NT = K / 32;
  const bool issuer = (wave == 0);

  if (issuer) {                       // prime the pipeline: tiles for kt=0
    tdm_load_2d(0u,     Abase, 32, 128, K, K, M);
    tdm_load_2d(16384u, Wbase, 32, 128, K, K, N);
  }

  v8f acc[4][2] = {};

  for (int kt = 0; kt < NT; ++kt) {
    const int cur = kt & 1;
    __bf16* curA = lds + (cur ? 4096 : 0);       // runtime select: no constant
    __bf16* curB = lds + (cur ? 12288 : 8192);   // aggregates of LDS pointers
    const unsigned nxtAOff = cur ? 0u : 8192u;
    const unsigned nxtBOff = cur ? 16384u : 24576u;

    if (issuer) {
      if (kt + 1 < NT) {              // prefetch next tiles into other buffer
        tdm_load_2d(nxtAOff, Abase + (kt + 1) * 32, 32, 128, K, K, M);
        tdm_load_2d(nxtBOff, Wbase + (kt + 1) * 32, 32, 128, K, K, N);
        __builtin_amdgcn_s_wait_tensorcnt((short)2);  // current pair complete
      } else {
        __builtin_amdgcn_s_wait_tensorcnt((short)0);
      }
    }
    __syncthreads();                  // tiles[cur] visible to all waves

    // A fragments: lane ll = row M, K = lh*8 + {0..7} and lh*8 + {16..23}
    Frag16 af[4];
#pragma unroll
    for (int mi = 0; mi < 4; ++mi) {
      const __bf16* base = curA + (wm * 64 + mi * 16 + ll) * 32 + lh * 8;
      af[mi].q[0] = *(const v4i*)(base);
      af[mi].q[1] = *(const v4i*)(base + 16);
    }
    // B fragments: B = W^T, lane ll = column N = W-row; K = lh*8 + {0..7} and lh*8 + {16..23}
    Frag16 bfr[2];
#pragma unroll
    for (int ni = 0; ni < 2; ++ni) {
      const __bf16* base = curB + (wn * 32 + ni * 16 + ll) * 32 + lh * 8;
      bfr[ni].q[0] = *(const v4i*)(base);
      bfr[ni].q[1] = *(const v4i*)(base + 16);
    }
#pragma unroll
    for (int mi = 0; mi < 4; ++mi)
#pragma unroll
      for (int ni = 0; ni < 2; ++ni)
        acc[mi][ni] = WMMA_BF16(af[mi].v, bfr[ni].v, acc[mi][ni]);
    __syncthreads();                  // done reading tiles[cur] before reuse
  }

  // Epilogue: C layout — VGPR r -> row r + 8*lh, col = ll
#pragma unroll
  for (int ni = 0; ni < 2; ++ni) {
    int gn = tileN + wn * 32 + ni * 16 + ll;
    float bb = bias[gn];
    int gq = gn / n_per_group;
    int gf = gn % n_per_group;
#pragma unroll
    for (int mi = 0; mi < 4; ++mi) {
#pragma unroll
      for (int r = 0; r < 8; ++r) {
        int gm = tileM + wm * 64 + mi * 16 + r + 8 * lh;
        float val = acc[mi][ni][r] + bb;
        long long off = (long long)gq * group_stride + (long long)gm * n_per_group + gf;
        if (OUT_F32) ((float*)outv)[off] = val;
        else         ((__bf16*)outv)[off] = (__bf16)val;
      }
    }
  }
}

// ---------------------------------------------------------------------------
// Fused per-token attention core. One wave per token m = b*S+s, looping Q=4:
//   scores(16x16) = q(16x64) . k(16x64)^T * scale   (2x WMMA, K=64)
//   softmax over columns (across 16-lane halves, __shfl_xor)
//   out(16x64)  += probs(16x16, zero-padded K) . v   (4x WMMA)
// Result (summed over Q) stored bf16 into ctx with the torch
// transpose(1,2)+view layout, which flattens to (B,H,S,D).
// ---------------------------------------------------------------------------
__global__ __launch_bounds__(128) void attn_core(
    const __bf16* __restrict__ qbuf,  // (Q, M, E)
    const __bf16* __restrict__ kbuf,  // (M, E)
    const __bf16* __restrict__ vbuf,  // (M, E)
    __bf16* __restrict__ ctx,         // (B, H, S, D) flat
    float scale) {
  __shared__ float probsLds[4 * 16 * 16];

  const int t    = threadIdx.x;
  const int wave = t >> 5;
  const int lane = t & 31;
  const int lh   = lane >> 4;
  const int ll   = lane & 15;
  const int m    = blockIdx.x * 4 + wave;
  const int b    = m / kS;
  const int s    = m % kS;

  const __bf16* krow = kbuf + (size_t)m * kE;
  const __bf16* vrow = vbuf + (size_t)m * kE;
  float* pl = probsLds + wave * 256;

  v8f oacc[4] = {};

  for (int qq = 0; qq < kQ; ++qq) {
    const __bf16* qrow = qbuf + ((size_t)qq * kM + m) * kE;

    // --- scores = q . k^T  (A = q rows, B = k^T => contiguous k-row slices)
    v8f sc = {};
#pragma unroll
    for (int c = 0; c < 2; ++c) {
      Frag16 a, bfr;
      const __bf16* ab = qrow + ll * kD + c * 32 + lh * 8;
      a.q[0] = *(const v4i*)(ab);
      a.q[1] = *(const v4i*)(ab + 16);
      const __bf16* bb = krow + ll * kD + c * 32 + lh * 8;
      bfr.q[0] = *(const v4i*)(bb);
      bfr.q[1] = *(const v4i*)(bb + 16);
      sc = WMMA_BF16(a.v, bfr.v, sc);
    }

    // --- softmax over columns g (N axis: across the 16 lanes of each half)
    float p[8];
#pragma unroll
    for (int r = 0; r < 8; ++r) {
      float x  = sc[r] * scale;
      float mx = x;
#pragma unroll
      for (int off = 8; off >= 1; off >>= 1)
        mx = fmaxf(mx, __shfl_xor(mx, off, 32));
      float e   = __expf(x - mx);
      float sum = e;
#pragma unroll
      for (int off = 8; off >= 1; off >>= 1)
        sum += __shfl_xor(sum, off, 32);
      p[r] = e / sum;
    }

    // --- relayout probs C(16x16) -> A-fragment via LDS; zero-pad K=16..31
#pragma unroll
    for (int r = 0; r < 8; ++r)
      pl[(r + 8 * lh) * 16 + ll] = p[r];
    __syncthreads();

    Frag16 pa;
#pragma unroll
    for (int j = 0; j < 8; ++j)
      pa.e[j] = (__bf16)pl[ll * 16 + lh * 8 + j];
#pragma unroll
    for (int j = 8; j < 16; ++j)
      pa.e[j] = (__bf16)0.0f;

    // --- out += probs . v   (B-frag: lane ll = col d, K rows g strided by D;
    //     K=16..31 half zeroed to match zero-padded A)
#pragma unroll
    for (int nb = 0; nb < 4; ++nb) {
      Frag16 vb;
#pragma unroll
      for (int j = 0; j < 16; ++j)
        vb.e[j] = (j < 8) ? vrow[(lh * 8 + j) * kD + nb * 16 + ll] : (__bf16)0.0f;
      oacc[nb] = WMMA_BF16(pa.v, vb.v, oacc[nb]);
    }
    __syncthreads();
  }

  // --- store summed context, layout (B,H,S,D) == torch transpose+view flat
#pragma unroll
  for (int nb = 0; nb < 4; ++nb) {
#pragma unroll
    for (int r = 0; r < 8; ++r) {
      int h = r + 8 * lh;
      int d = nb * 16 + ll;
      ctx[(((size_t)b * kH + h) * kS + s) * kD + d] = (__bf16)oacc[nb][r];
    }
  }
}

// ---------------------------------------------------------------------------
// Host-side orchestration (all on `stream`, graph-capture safe).
// ---------------------------------------------------------------------------
extern "C" void kernel_launch(void* const* d_in, const int* in_sizes, int n_in,
                              void* d_out, int out_size, void* d_ws, size_t ws_size,
                              hipStream_t stream) {
  (void)in_sizes; (void)n_in; (void)out_size; (void)ws_size;
  const float* x  = (const float*)d_in[0];
  const float* Wq = (const float*)d_in[1];
  const float* bq = (const float*)d_in[2];
  const float* Wk = (const float*)d_in[3];
  const float* bk = (const float*)d_in[4];
  const float* Wv = (const float*)d_in[5];
  const float* bv = (const float*)d_in[6];
  const float* Wo = (const float*)d_in[7];
  const float* bo = (const float*)d_in[8];

  __bf16* ws = (__bf16*)d_ws;
  size_t o = 0;
  __bf16* xbf  = ws + o; o += (size_t)kM * kE;        // 8M
  __bf16* wqbf = ws + o; o += (size_t)kQ * kE * kE;   // 4M
  __bf16* wkbf = ws + o; o += (size_t)kE * kE;        // 1M
  __bf16* wvbf = ws + o; o += (size_t)kE * kE;        // 1M
  __bf16* wobf = ws + o; o += (size_t)kE * kE;        // 1M
  __bf16* qbuf = ws + o; o += (size_t)kQ * kM * kE;   // 32M  (Q, M, E)
  __bf16* kbuf = ws + o; o += (size_t)kM * kE;        // 8M
  __bf16* vbuf = ws + o; o += (size_t)kM * kE;        // 8M
  __bf16* ctx  = ws + o; o += (size_t)kM * kE;        // 8M   (B,H,S,D)

  auto cvt = [&](const float* src, __bf16* dst, int n) {
    cvt_f32_bf16<<<n / (256 * 4), 256, 0, stream>>>(src, dst, n);
  };
  cvt(x,  xbf,  kM * kE);
  cvt(Wq, wqbf, kQ * kE * kE);
  cvt(Wk, wkbf, kE * kE);
  cvt(Wv, wvbf, kE * kE);
  cvt(Wo, wobf, kE * kE);

  // Q projections (stacked along N; output grouped into (Q, M, E))
  gemm_bf16_tn<false><<<dim3((kQ * kE) / 128, kM / 128), 256, 0, stream>>>(
      xbf, wqbf, bq, qbuf, kM, kQ * kE, kE, kE, (long long)kM * kE);
  // K / V projections
  gemm_bf16_tn<false><<<dim3(kE / 128, kM / 128), 256, 0, stream>>>(
      xbf, wkbf, bk, kbuf, kM, kE, kE, kE, 0LL);
  gemm_bf16_tn<false><<<dim3(kE / 128, kM / 128), 256, 0, stream>>>(
      xbf, wvbf, bv, vbuf, kM, kE, kE, kE, 0LL);

  // Per-token head-mixing attention, summed over Q, stored as (B,H,S,D) bf16
  attn_core<<<kM / 4, 128, 0, stream>>>(qbuf, kbuf, vbuf, ctx, 0.125f);

  // Output projection -> d_out (fp32)
  gemm_bf16_tn<true><<<dim3(kE / 128, kM / 128), 256, 0, stream>>>(
      ctx, wobf, bo, d_out, kM, kE, kE, kE, 0LL);
}
